// GENI_4045859193726
// MI455X (gfx1250) — compile-verified, matrix operand order checked by reading, not run on hardware
//
#include <hip/hip_runtime.h>

#define NN    6144
#define ADIM  128
#define HIDN  96
#define EPT   100000
#define NTYPES 3
#define NE    (NTYPES*EPT)
#define MAXROW 2048

typedef __attribute__((ext_vector_type(2)))  float    v2f;
typedef __attribute__((ext_vector_type(8)))  float    v8f;
typedef __attribute__((ext_vector_type(16))) _Float16 v16h;

__device__ __forceinline__ float leaky(float x){ return x > 0.f ? x : 0.2f * x; }

// ---------------- CSR build (graph structure is layer/head invariant) -------

__global__ void zero_ints(int* __restrict__ p, int n){
  int i = blockIdx.x * blockDim.x + threadIdx.x;
  if (i < n) p[i] = 0;
}

__global__ void count_deg(const int* __restrict__ edges, int* __restrict__ deg){
  int g = blockIdx.x * blockDim.x + threadIdx.x;
  if (g >= NE) return;
  int t = g / EPT, e = g % EPT;
  int src = edges[t * 2 * EPT + e];
  atomicAdd(&deg[src], 1);
}

__global__ void scan_deg(const int* __restrict__ deg, int* __restrict__ row_start){
  __shared__ int part[256];
  const int CH = NN / 256;            // 24
  int tid = threadIdx.x, base = tid * CH, sum = 0;
  for (int i = 0; i < CH; i++) sum += deg[base + i];
  part[tid] = sum;
  __syncthreads();
  if (tid == 0){
    int acc = 0;
    for (int i = 0; i < 256; i++){ int v = part[i]; part[i] = acc; acc += v; }
  }
  __syncthreads();
  int acc = part[tid];
  for (int i = 0; i < CH; i++){ row_start[base + i] = acc; acc += deg[base + i]; }
  if (tid == 255) row_start[NN] = acc;
}

__global__ void scatter_edges(const int* __restrict__ edges, const int* __restrict__ row_start,
                              int* __restrict__ cursor, int* __restrict__ entry_pk){
  int g = blockIdx.x * blockDim.x + threadIdx.x;
  if (g >= NE) return;
  int t = g / EPT, e = g % EPT;
  int src = edges[t * 2 * EPT + e];
  int tgt = edges[t * 2 * EPT + EPT + e];
  int pos = row_start[src] + atomicAdd(&cursor[src], 1);
  entry_pk[pos] = (tgt << 2) | t;     // sortable key: (col, type)
}

// Per-row: sort keys (canonical order -> deterministic FP sums), merge duplicate
// columns into (col, n0, n1, n2).
__global__ void dedup_rows(const int* __restrict__ deg, const int* __restrict__ row_start,
                           const int* __restrict__ entry_pk,
                           int* __restrict__ ded_col, unsigned* __restrict__ ded_cnt,
                           int* __restrict__ ded_deg){
  __shared__ int key[MAXROW];
  int row = blockIdx.x;
  int k = deg[row]; if (k > MAXROW) k = MAXROW;   // binomial max ~100; clamp is pure safety
  int base = row_start[row];
  for (int j = threadIdx.x; j < k; j += blockDim.x) key[j] = entry_pk[base + j];
  __syncthreads();
  for (int phase = 0; phase < k; phase++){        // odd-even transposition sort
    int start = phase & 1;
    for (int i = start + 2 * (int)threadIdx.x; i + 1 < k; i += 2 * (int)blockDim.x){
      int a = key[i], b = key[i + 1];
      if (a > b){ key[i] = b; key[i + 1] = a; }
    }
    __syncthreads();
  }
  if (threadIdx.x == 0){
    int out = 0, j = 0;
    while (j < k){
      int c = key[j] >> 2;
      int n0 = 0, n1 = 0, n2 = 0;
      while (j < k && (key[j] >> 2) == c){
        int t = key[j] & 3;
        n0 += (t == 0); n1 += (t == 1); n2 += (t == 2);
        j++;
      }
      ded_col[base + out] = c;
      ded_cnt[base + out] = (unsigned)n0 | ((unsigned)n1 << 10) | ((unsigned)n2 << 20);
      out++;
    }
    ded_deg[row] = out;
  }
}

// ---------------- score GEMM via WMMA ---------------------------------------
// One wave computes a 16-row tile:  H = leaky(X[16x128] @ W1[128x96] + b1),
// then s = H @ W2 + b2.  96 = 6 N-tiles of 16; 128 = 32 K-steps of 4 (f32 path).

#if defined(__has_builtin)
#if __has_builtin(__builtin_amdgcn_wmma_f32_16x16x4_f32)
#define USE_WMMA_F32 1
#endif
#endif

__global__ __launch_bounds__(32)
void score_gemm(const float* __restrict__ X, const float* __restrict__ W1,
                const float* __restrict__ b1, const float* __restrict__ W2,
                const float* __restrict__ b2, float* __restrict__ s){
  const int lane = threadIdx.x;
  const int nlo  = lane & 15;
  const int hi   = lane >> 4;       // 0: K {0,1} half, 1: K {2,3} half
  const int m0   = blockIdx.x << 4;
  float partial[8];
#pragma unroll
  for (int r = 0; r < 8; r++) partial[r] = 0.f;

#if USE_WMMA_F32
  // A 16x4 f32 layout: lanes 0-15 M=0..15 hold K=k0,k0+1 ; lanes 16-31 hold K=k0+2,k0+3
  v2f a[32];
  const float* xr = X + (m0 + nlo) * ADIM + hi * 2;
#pragma unroll
  for (int k = 0; k < 32; k++) a[k] = *(const v2f*)(xr + 4 * k);
  for (int t = 0; t < 6; t++){
    float bias = b1[t * 16 + nlo];
    v8f c;
#pragma unroll
    for (int r = 0; r < 8; r++) c[r] = bias;
    const float* wc = W1 + (hi * 2) * HIDN + t * 16 + nlo;   // B 4x16: n=nlo, K split by half
#pragma unroll
    for (int k = 0; k < 32; k++){
      v2f b;
      b.x = wc[(4 * k) * HIDN];
      b.y = wc[(4 * k + 1) * HIDN];
      c = __builtin_amdgcn_wmma_f32_16x16x4_f32(false, a[k], false, b, (short)0, c, false, false);
    }
    float w2v = W2[t * 16 + nlo];
#pragma unroll
    for (int r = 0; r < 8; r++) partial[r] += leaky(c[r]) * w2v;   // C/D: m = r + 8*hi, n = nlo
  }
#else
  // Fallback: confirmed f16 WMMA, converting on the fly.
  // 16-bit A 16x32 layout: element e of v16h holds K = (e/8)*16 + hi*8 + (e%8)
  v16h a[4];
  const float* xr = X + (m0 + nlo) * ADIM;
#pragma unroll
  for (int kc = 0; kc < 4; kc++)
#pragma unroll
    for (int e = 0; e < 16; e++){
      int kk = kc * 32 + (e >> 3) * 16 + hi * 8 + (e & 7);
      a[kc][e] = (_Float16)xr[kk];
    }
  for (int t = 0; t < 6; t++){
    float bias = b1[t * 16 + nlo];
    v8f c;
#pragma unroll
    for (int r = 0; r < 8; r++) c[r] = bias;
#pragma unroll
    for (int kc = 0; kc < 4; kc++){
      v16h b;
#pragma unroll
      for (int e = 0; e < 16; e++){
        int kk = kc * 32 + (e >> 3) * 16 + hi * 8 + (e & 7);
        b[e] = (_Float16)W1[kk * HIDN + t * 16 + nlo];
      }
      c = __builtin_amdgcn_wmma_f32_16x16x32_f16(false, a[kc], false, b, (short)0, c, false, false);
    }
    float w2v = W2[t * 16 + nlo];
#pragma unroll
    for (int r = 0; r < 8; r++) partial[r] += leaky(c[r]) * w2v;
  }
#endif
  // reduce over the 16 n-lanes inside each half (halves hold different m rows)
#pragma unroll
  for (int r = 0; r < 8; r++){
    float v = partial[r];
    v += __shfl_xor(v, 1, 32);
    v += __shfl_xor(v, 2, 32);
    v += __shfl_xor(v, 4, 32);
    v += __shfl_xor(v, 8, 32);
    partial[r] = v;
  }
  if (nlo == 0){
    float bb = b2[0];
#pragma unroll
    for (int r = 0; r < 8; r++) s[m0 + 8 * hi + r] = partial[r] + bb;
  }
}

// ---------------- per-layer sparse softmax-matvec ---------------------------

__global__ void reduce_sum(const float* __restrict__ s, float* __restrict__ out){
  __shared__ float p[256];
  float acc = 0.f;
  for (int i = threadIdx.x; i < NN; i += 256) acc += s[i];
  p[threadIdx.x] = acc;
  __syncthreads();
  for (int st = 128; st > 0; st >>= 1){
    if ((int)threadIdx.x < st) p[threadIdx.x] += p[threadIdx.x + st];
    __syncthreads();
  }
  if (threadIdx.x == 0) *out = p[0];
}

__global__ __launch_bounds__(32)
void layer_pass(const float* __restrict__ s, const int* __restrict__ row_start,
                const int* __restrict__ ded_deg, const int* __restrict__ ded_col,
                const unsigned* __restrict__ ded_cnt,
                const float* __restrict__ emb, const float* __restrict__ aw,
                int layer, const float* __restrict__ stot, float* __restrict__ s_out){
  int row = blockIdx.x;
  int lane = threadIdx.x;
  float w0[2], wl[2], es[2][3];
#pragma unroll
  for (int h = 0; h < 2; h++){
    const float* w = aw + (layer * 2 + h) * 22;     // attn_w[l,h,:,0], len 22
    w0[h] = w[0]; wl[h] = w[21];
#pragma unroll
    for (int t = 0; t < 3; t++){
      float acc = 0.f;
      for (int d = 0; d < 20; d++) acc += emb[t * 20 + d] * w[1 + d];
      es[h][t] = acc;
    }
  }
  float si = s[row];
  int base = row_start[row];
  int k = ded_deg[row];
  float den0 = 0.f, num0 = 0.f, den1 = 0.f, num1 = 0.f;
  for (int j = lane; j < k; j += 32){
    int c = ded_col[base + j];
    unsigned pc = ded_cnt[base + j];
    float n0 = (float)(pc & 1023u);
    float n1 = (float)((pc >> 10) & 1023u);
    float n2 = (float)((pc >> 20) & 1023u);
    float cnt = n0 + n1 + n2;
    float sj = s[c];
    float a0 = cnt * (w0[0] * si + wl[0] * sj) + n0 * es[0][0] + n1 * es[0][1] + n2 * es[0][2];
    float v0 = __expf(leaky(a0)) - 1.0f;
    den0 += v0; num0 += v0 * sj;
    float a1 = cnt * (w0[1] * si + wl[1] * sj) + n0 * es[1][0] + n1 * es[1][1] + n2 * es[1][2];
    float v1 = __expf(leaky(a1)) - 1.0f;
    den1 += v1; num1 += v1 * sj;
  }
#pragma unroll
  for (int m = 16; m >= 1; m >>= 1){
    den0 += __shfl_xor(den0, m, 32);
    num0 += __shfl_xor(num0, m, 32);
    den1 += __shfl_xor(den1, m, 32);
    num1 += __shfl_xor(num1, m, 32);
  }
  if (lane == 0){
    float S = *stot;
    float o0 = (S + num0) / ((float)NN + den0);
    float o1 = (S + num1) / ((float)NN + den1);
    s_out[row] = 0.5f * (o0 + o1);      // acc / NUM_HEADS
  }
}

// ---------------- launch ----------------------------------------------------

extern "C" void kernel_launch(void* const* d_in, const int* in_sizes, int n_in,
                              void* d_out, int out_size, void* d_ws, size_t ws_size,
                              hipStream_t stream){
  const float* X     = (const float*)d_in[0];
  const int*   edges = (const int*)d_in[1];   // JAX default: int64 request silently -> int32
  const float* W1    = (const float*)d_in[2];
  const float* b1    = (const float*)d_in[3];
  const float* W2    = (const float*)d_in[4];
  const float* b2    = (const float*)d_in[5];
  const float* emb   = (const float*)d_in[6];
  const float* aw    = (const float*)d_in[7];
  float* out = (float*)d_out;

  int* deg        = (int*)d_ws;              // N
  int* cursor     = deg + NN;                // N
  int* row_start  = cursor + NN;             // N+1
  int* ded_deg    = row_start + (NN + 1);    // N
  int* entry_pk   = ded_deg + NN;            // 3E
  int* ded_col    = entry_pk + NE;           // 3E
  unsigned* ded_cnt = (unsigned*)(ded_col + NE); // 3E
  float* s0   = (float*)(ded_cnt + NE);      // N
  float* s1   = s0 + NN;                     // N
  float* stot = s1 + NN;                     // 1     (total ~3.75 MB)

  zero_ints    <<<(2 * NN + 255) / 256, 256, 0, stream>>>(deg, 2 * NN); // deg + cursor
  count_deg    <<<(NE + 255) / 256,     256, 0, stream>>>(edges, deg);
  scan_deg     <<<1,                    256, 0, stream>>>(deg, row_start);
  scatter_edges<<<(NE + 255) / 256,     256, 0, stream>>>(edges, row_start, cursor, entry_pk);
  dedup_rows   <<<NN,                   128, 0, stream>>>(deg, row_start, entry_pk,
                                                          ded_col, ded_cnt, ded_deg);
  score_gemm   <<<NN / 16,               32, 0, stream>>>(X, W1, b1, W2, b2, s0);

  reduce_sum   <<<1, 256, 0, stream>>>(s0, stot);
  layer_pass   <<<NN, 32, 0, stream>>>(s0, row_start, ded_deg, ded_col, ded_cnt,
                                       emb, aw, 0, stot, s1);
  reduce_sum   <<<1, 256, 0, stream>>>(s1, stot);
  layer_pass   <<<NN, 32, 0, stream>>>(s1, row_start, ded_deg, ded_col, ded_cnt,
                                       emb, aw, 1, stot, out);
}
